// _RelationMessagePassing_79121887527265
// MI455X (gfx1250) — matrix operand order repaired and checked by reading, not compiled, not run on hardware
//
#include <hip/hip_runtime.h>
#include <hip/hip_fp16.h>

typedef __attribute__((ext_vector_type(16))) _Float16 v16h;
typedef __attribute__((ext_vector_type(8)))  _Float16 v8h;
typedef __attribute__((ext_vector_type(8)))  float    v8f;

#define ALPHA_F 8.0f
#define INV_ALPHA_F 0.125f

// ---------- order-preserving float <-> uint for atomic max ----------
__device__ __forceinline__ unsigned f2ord(float f) {
  unsigned u = __float_as_uint(f);
  return (u & 0x80000000u) ? ~u : (u | 0x80000000u);
}
__device__ __forceinline__ float ord2f(unsigned k) {
  return __uint_as_float((k & 0x80000000u) ? (k & 0x7fffffffu) : ~k);
}

// ---------- WMMA helpers (CDNA5 wave32, 16x16x32 f16 -> f32) ----------
__device__ __forceinline__ v8f wmma16(v16h a, v16h b, v8f c) {
  return __builtin_amdgcn_wmma_f32_16x16x32_f16(false, a, false, b, (short)0, c,
                                                false, false);
}

// A fragment: rows M=lane&15 of a [16][ld] f16 tile, K window [k0, k0+32).
// Layout (ISA 7.12.2): lanes 0-15 halves 0-7 = K k0..k0+7, halves 8-15 = K k0+16..k0+23;
// lanes 16-31 shifted by +8.
__device__ __forceinline__ v16h load_a_frag(const _Float16* tile, int ld, int k0) {
  int lane = threadIdx.x & 31;
  int m = lane & 15;
  int kb = k0 + ((lane & 16) ? 8 : 0);
  const v8h* p0 = (const v8h*)(tile + (size_t)m * ld + kb);
  const v8h* p1 = (const v8h*)(tile + (size_t)m * ld + kb + 16);
  v8h lo = *p0, hi = *p1;
  v16h r;
#pragma unroll
  for (int i = 0; i < 8; ++i) { r[i] = lo[i]; r[i + 8] = hi[i]; }
  return r;
}

// B fragment from TRANSPOSED weights Wt[n][k] (row stride ld halves):
// lane n<16: halves j<8 = Wt[n][k0+j], j>=8 = Wt[n][k0+16+(j-8)]
// lane>=16:  same, shifted by +8 in K
__device__ __forceinline__ v16h load_b_frag(const _Float16* wt, int ld, int n0, int k0) {
  int lane = threadIdx.x & 31;
  int n = n0 + (lane & 15);
  int kb = k0 + ((lane & 16) ? 8 : 0);
  const v8h* p = (const v8h*)(wt + (size_t)n * ld + kb);
  v8h lo = p[0], hi = p[2];
  v16h r;
#pragma unroll
  for (int i = 0; i < 8; ++i) { r[i] = lo[i]; r[i + 8] = hi[i]; }
  return r;
}

// ---------- small prep kernels ----------
__global__ void k_f32_to_f16(const float* __restrict__ src, _Float16* __restrict__ dst,
                             long n) {
  size_t i = (size_t)blockIdx.x * 256 + threadIdx.x;
  if (i < (size_t)n) dst[i] = (_Float16)src[i];
}

// src is [K][Nn] fp32 (weights stored [in,out]); dst is [Nn][K] f16 (transposed)
__global__ void k_transpose_f16(const float* __restrict__ src, _Float16* __restrict__ dst,
                                int K, int Nn) {
  size_t i = (size_t)blockIdx.x * 256 + threadIdx.x;
  if (i < (size_t)K * Nn) {
    int k = (int)(i / Nn);
    int n = (int)(i - (size_t)k * Nn);
    dst[(size_t)n * K + k] = (_Float16)src[i];
  }
}

__global__ void k_init(float* __restrict__ exps, unsigned* __restrict__ gmaxbits,
                       float* __restrict__ gsum, unsigned* __restrict__ msgmaxbits,
                       long n) {
  size_t i = (size_t)blockIdx.x * 256 + threadIdx.x;
  if (i < (size_t)n) exps[i] = 1e-16f;
  if (i < 64) { gmaxbits[i] = 0u; gsum[i] = 0.0f; }
  if (i == 0) msgmaxbits[0] = 0u;
}

// column max over node_states [N][64]
__global__ void k_colmax(const float* __restrict__ ns, unsigned* __restrict__ gmax, int N) {
  __shared__ float red[256];
  int col = threadIdx.x & 63, grp = threadIdx.x >> 6;
  float m = -3.402823466e38f;
  for (int r = blockIdx.x * 4 + grp; r < N; r += gridDim.x * 4)
    m = fmaxf(m, ns[(size_t)r * 64 + col]);
  red[threadIdx.x] = m;
  __syncthreads();
  if (grp == 0) {
    m = fmaxf(fmaxf(red[col], red[64 + col]), fmaxf(red[128 + col], red[192 + col]));
    atomicMax(&gmax[col], f2ord(m));
  }
}

// column sum of exp(alpha*(x - colmax))
__global__ void k_colexpsum(const float* __restrict__ ns, const unsigned* __restrict__ gmax,
                            float* __restrict__ gsum, int N) {
  __shared__ float red[256];
  int col = threadIdx.x & 63, grp = threadIdx.x >> 6;
  float off = ord2f(gmax[col]);
  float s = 0.0f;
  for (int r = blockIdx.x * 4 + grp; r < N; r += gridDim.x * 4)
    s += expf(ALPHA_F * (ns[(size_t)r * 64 + col] - off));
  red[threadIdx.x] = s;
  __syncthreads();
  if (grp == 0) {
    s = red[col] + red[64 + col] + red[128 + col] + red[192 + col];
    atomicAdd(&gsum[col], s);
  }
}

__global__ void k_gemb(const unsigned* __restrict__ gmax, const float* __restrict__ gsum,
                       float* __restrict__ gemb) {
  int i = threadIdx.x;
  if (i < 64) gemb[i] = logf(gsum[i]) * INV_ALPHA_F + ord2f(gmax[i]);
}

// ---------- relation MLP: [T,D] gathered -> Linear(D,D)+ReLU -> Linear(D,D) ----------
// 4 waves / block; each wave computes a 16-tuple x D stripe with WMMA.
// A-fragments are hoisted into registers and reused across all D/16 output tiles.
template <int D, int A>
__global__ __launch_bounds__(128) void k_rel_mlp(
    const _Float16* __restrict__ nodes16, const int* __restrict__ idx,
    const _Float16* __restrict__ wt1, const float* __restrict__ b1,
    const _Float16* __restrict__ wt2, const float* __restrict__ b2,
    float* __restrict__ Y, unsigned* __restrict__ maxbits, int T) {
  __shared__ alignas(16) _Float16 At[4][16 * D];
  __shared__ alignas(16) _Float16 Hm[4][16 * D];
  __shared__ float red[128];
  int w = threadIdx.x >> 5, lane = threadIdx.x & 31;
  int t0 = (blockIdx.x * 4 + w) * 16;
  _Float16* at = At[w];
  _Float16* hm = Hm[w];

  // gather: row t = concat of A node states (64 f16 = 128B each, 16B copies)
  for (int i = lane; i < 16 * A; i += 32) {
    int row = i / A, s = i - row * A;
    int node = idx[(size_t)(t0 + row) * A + s];
    const uint4* src = (const uint4*)(nodes16 + (size_t)node * 64);
    uint4* dst = (uint4*)(at + row * D + s * 64);
#pragma unroll
    for (int j = 0; j < 8; ++j) dst[j] = src[j];
  }
  __syncthreads();

  // GEMM1: H = relu(X @ W1 + b1), stored f16 back to LDS
  {
    v16h af[D / 32];
#pragma unroll
    for (int kk = 0; kk < D / 32; ++kk) af[kk] = load_a_frag(at, D, kk * 32);
#pragma unroll
    for (int nt = 0; nt < D / 16; ++nt) {
      float bv = b1[nt * 16 + (lane & 15)];
      v8f acc;
#pragma unroll
      for (int r = 0; r < 8; ++r) acc[r] = bv;
#pragma unroll
      for (int kk = 0; kk < D / 32; ++kk)
        acc = wmma16(af[kk], load_b_frag(wt1, D, nt * 16, kk * 32), acc);
      int col = nt * 16 + (lane & 15);
      int rb = (lane & 16) ? 8 : 0;
#pragma unroll
      for (int r = 0; r < 8; ++r) hm[(rb + r) * D + col] = (_Float16)fmaxf(acc[r], 0.0f);
    }
  }
  __syncthreads();

  // GEMM2: Y = H @ W2 + b2 -> global fp32, track max
  float tmax = -3.402823466e38f;
  {
    v16h af[D / 32];
#pragma unroll
    for (int kk = 0; kk < D / 32; ++kk) af[kk] = load_a_frag(hm, D, kk * 32);
#pragma unroll
    for (int nt = 0; nt < D / 16; ++nt) {
      float bv = b2[nt * 16 + (lane & 15)];
      v8f acc;
#pragma unroll
      for (int r = 0; r < 8; ++r) acc[r] = bv;
#pragma unroll
      for (int kk = 0; kk < D / 32; ++kk)
        acc = wmma16(af[kk], load_b_frag(wt2, D, nt * 16, kk * 32), acc);
      int col = nt * 16 + (lane & 15);
      int rb = (lane & 16) ? 8 : 0;
#pragma unroll
      for (int r = 0; r < 8; ++r) {
        float v = acc[r];
        Y[(size_t)(t0 + rb + r) * D + col] = v;
        tmax = fmaxf(tmax, v);
      }
    }
  }

  // block max -> global ordered-int atomic max
  red[threadIdx.x] = tmax;
  __syncthreads();
  if (threadIdx.x < 32) {
    float m = fmaxf(fmaxf(red[threadIdx.x], red[threadIdx.x + 32]),
                    fmaxf(red[threadIdx.x + 64], red[threadIdx.x + 96]));
#pragma unroll
    for (int o = 16; o > 0; o >>= 1) m = fmaxf(m, __shfl_down(m, o, 32));
    if (threadIdx.x == 0) atomicMax(maxbits, f2ord(m));
  }
}

// ---------- scatter: exps[node][h] += exp(alpha*(y - maxoff)) ----------
__global__ void k_scatter(const float* __restrict__ Y, const int* __restrict__ idx,
                          const unsigned* __restrict__ maxbits, float* __restrict__ exps,
                          long total, int D, int A) {
  size_t i = (size_t)blockIdx.x * 256 + threadIdx.x;
  if (i >= (size_t)total) return;
  int t = (int)(i / D);
  int j = (int)(i - (size_t)t * D);
  int s = j >> 6, h = j & 63;
  int node = idx[(size_t)t * A + s];
  float mo = ord2f(*maxbits);
  atomicAdd(&exps[(size_t)node * 64 + h], expf(ALPHA_F * (Y[i] - mo)));
}

// ---------- final update MLP: [N,224] -> 128 (relu) -> 64 ----------
__global__ __launch_bounds__(128) void k_update(
    const _Float16* __restrict__ nodes16, const float* __restrict__ extra,
    const float* __restrict__ gemb, const float* __restrict__ exps,
    const unsigned* __restrict__ maxbits, const _Float16* __restrict__ ut1,
    const float* __restrict__ ub1, const _Float16* __restrict__ ut2,
    const float* __restrict__ ub2, float* __restrict__ out, int N) {
  const int K1 = 224, DM = 128, DO = 64;
  __shared__ alignas(16) _Float16 F[4][16 * 224];
  __shared__ alignas(16) _Float16 Hm[4][16 * 128];
  int w = threadIdx.x >> 5, lane = threadIdx.x & 31;
  int n0 = (blockIdx.x * 4 + w) * 16;
  _Float16* f = F[w];
  _Float16* hm = Hm[w];
  float mo = ord2f(*maxbits);

  // feature tile: [extra(32) | graph_emb(64) | messages(64) | node(64)]
  for (int i = lane; i < 16 * 32; i += 32) {
    int row = i >> 5, c = i & 31;
    f[row * K1 + c] = (_Float16)extra[c];
  }
  for (int i = lane; i < 16 * 64; i += 32) {
    int row = i >> 6, c = i & 63;
    f[row * K1 + 32 + c] = (_Float16)gemb[c];
  }
  for (int i = lane; i < 16 * 64; i += 32) {
    int row = i >> 6, c = i & 63;
    float e = exps[(size_t)(n0 + row) * 64 + c];
    f[row * K1 + 96 + c] = (_Float16)(logf(e) * INV_ALPHA_F + mo);
  }
  for (int i = lane; i < 16 * 64; i += 32) {
    int row = i >> 6, c = i & 63;
    f[row * K1 + 160 + c] = nodes16[(size_t)(n0 + row) * 64 + c];
  }
  __syncthreads();

  // GEMM1: 224 -> 128, relu
  {
    v16h af[K1 / 32];
#pragma unroll
    for (int kk = 0; kk < K1 / 32; ++kk) af[kk] = load_a_frag(f, K1, kk * 32);
#pragma unroll
    for (int nt = 0; nt < DM / 16; ++nt) {
      float bv = ub1[nt * 16 + (lane & 15)];
      v8f acc;
#pragma unroll
      for (int r = 0; r < 8; ++r) acc[r] = bv;
#pragma unroll
      for (int kk = 0; kk < K1 / 32; ++kk)
        acc = wmma16(af[kk], load_b_frag(ut1, K1, nt * 16, kk * 32), acc);
      int col = nt * 16 + (lane & 15);
      int rb = (lane & 16) ? 8 : 0;
#pragma unroll
      for (int r = 0; r < 8; ++r) hm[(rb + r) * DM + col] = (_Float16)fmaxf(acc[r], 0.0f);
    }
  }
  __syncthreads();

  // GEMM2: 128 -> 64
  {
    v16h af[DM / 32];
#pragma unroll
    for (int kk = 0; kk < DM / 32; ++kk) af[kk] = load_a_frag(hm, DM, kk * 32);
#pragma unroll
    for (int nt = 0; nt < DO / 16; ++nt) {
      float bv = ub2[nt * 16 + (lane & 15)];
      v8f acc;
#pragma unroll
      for (int r = 0; r < 8; ++r) acc[r] = bv;
#pragma unroll
      for (int kk = 0; kk < DM / 32; ++kk)
        acc = wmma16(af[kk], load_b_frag(ut2, DM, nt * 16, kk * 32), acc);
      int col = nt * 16 + (lane & 15);
      int rb = (lane & 16) ? 8 : 0;
#pragma unroll
      for (int r = 0; r < 8; ++r)
        out[(size_t)(n0 + rb + r) * 64 + nt * 16 + col] = acc[r];
    }
  }
}

// ---------- host side ----------
extern "C" void kernel_launch(void* const* d_in, const int* in_sizes, int n_in,
                              void* d_out, int out_size, void* d_ws, size_t ws_size,
                              hipStream_t stream) {
  const float* ns = (const float*)d_in[0];
  const float* extra = (const float*)d_in[1];
  const int* idxr[4] = {(const int*)d_in[2], (const int*)d_in[3],
                        (const int*)d_in[4], (const int*)d_in[5]};
  const float* W1[4] = {(const float*)d_in[6], (const float*)d_in[10],
                        (const float*)d_in[14], (const float*)d_in[18]};
  const float* B1[4] = {(const float*)d_in[7], (const float*)d_in[11],
                        (const float*)d_in[15], (const float*)d_in[19]};
  const float* W2[4] = {(const float*)d_in[8], (const float*)d_in[12],
                        (const float*)d_in[16], (const float*)d_in[20]};
  const float* B2[4] = {(const float*)d_in[9], (const float*)d_in[13],
                        (const float*)d_in[17], (const float*)d_in[21]};
  const float* uW1 = (const float*)d_in[22];
  const float* ub1 = (const float*)d_in[23];
  const float* uW2 = (const float*)d_in[24];
  const float* ub2 = (const float*)d_in[25];

  const int N = in_sizes[0] / 64;
  const int Tr[4] = {in_sizes[2], in_sizes[3] / 2, in_sizes[4] / 2, in_sizes[5] / 3};
  const int Dd[4] = {64, 128, 128, 192};
  const int Ar[4] = {1, 2, 2, 3};

  // workspace carve (256B aligned)
  char* ws = (char*)d_ws;
  size_t off = 0;
  auto carve = [&](size_t bytes) -> char* {
    char* p = ws + off;
    off += (bytes + 255) & ~(size_t)255;
    return p;
  };
  _Float16* nodes16 = (_Float16*)carve((size_t)N * 64 * 2);
  _Float16* wt1[4];
  _Float16* wt2[4];
  for (int r = 0; r < 4; ++r) {
    wt1[r] = (_Float16*)carve((size_t)Dd[r] * Dd[r] * 2);
    wt2[r] = (_Float16*)carve((size_t)Dd[r] * Dd[r] * 2);
  }
  _Float16* ut1 = (_Float16*)carve((size_t)128 * 224 * 2);
  _Float16* ut2 = (_Float16*)carve((size_t)64 * 128 * 2);
  unsigned* gmaxbits = (unsigned*)carve(64 * 4);
  float* gsum = (float*)carve(64 * 4);
  float* gemb = (float*)carve(64 * 4);
  unsigned* msgmaxbits = (unsigned*)carve(4);
  float* Yr[4];
  for (int r = 0; r < 4; ++r) Yr[r] = (float*)carve((size_t)Tr[r] * Dd[r] * 4);
  float* exps = (float*)carve((size_t)N * 64 * 4);
  (void)ws_size; (void)n_in; (void)out_size;

  // prep
  long nn = (long)N * 64;
  k_f32_to_f16<<<(int)((nn + 255) / 256), 256, 0, stream>>>(ns, nodes16, nn);
  for (int r = 0; r < 4; ++r) {
    long e = (long)Dd[r] * Dd[r];
    k_transpose_f16<<<(int)((e + 255) / 256), 256, 0, stream>>>(W1[r], wt1[r], Dd[r], Dd[r]);
    k_transpose_f16<<<(int)((e + 255) / 256), 256, 0, stream>>>(W2[r], wt2[r], Dd[r], Dd[r]);
  }
  k_transpose_f16<<<(224 * 128 + 255) / 256, 256, 0, stream>>>(uW1, ut1, 224, 128);
  k_transpose_f16<<<(128 * 64 + 255) / 256, 256, 0, stream>>>(uW2, ut2, 128, 64);
  k_init<<<(int)((nn + 255) / 256), 256, 0, stream>>>(exps, gmaxbits, gsum, msgmaxbits, nn);

  // graph embedding (smooth max over nodes)
  k_colmax<<<256, 256, 0, stream>>>(ns, gmaxbits, N);
  k_colexpsum<<<256, 256, 0, stream>>>(ns, gmaxbits, gsum, N);
  k_gemb<<<1, 64, 0, stream>>>(gmaxbits, gsum, gemb);

  // relation MLPs (WMMA); T divisible by 64 for this harness
  k_rel_mlp<64, 1><<<Tr[0] / 64, 128, 0, stream>>>(nodes16, idxr[0], wt1[0], B1[0],
                                                   wt2[0], B2[0], Yr[0], msgmaxbits, Tr[0]);
  k_rel_mlp<128, 2><<<Tr[1] / 64, 128, 0, stream>>>(nodes16, idxr[1], wt1[1], B1[1],
                                                    wt2[1], B2[1], Yr[1], msgmaxbits, Tr[1]);
  k_rel_mlp<128, 2><<<Tr[2] / 64, 128, 0, stream>>>(nodes16, idxr[2], wt1[2], B1[2],
                                                    wt2[2], B2[2], Yr[2], msgmaxbits, Tr[2]);
  k_rel_mlp<192, 3><<<Tr[3] / 64, 128, 0, stream>>>(nodes16, idxr[3], wt1[3], B1[3],
                                                    wt2[3], B2[3], Yr[3], msgmaxbits, Tr[3]);

  // scatter (after ALL relations: msg max must be global)
  for (int r = 0; r < 4; ++r) {
    long total = (long)Tr[r] * Dd[r];
    k_scatter<<<(int)((total + 255) / 256), 256, 0, stream>>>(Yr[r], idxr[r], msgmaxbits,
                                                              exps, total, Dd[r], Ar[r]);
  }

  // final update MLP (WMMA) -> d_out
  k_update<<<N / 64, 128, 0, stream>>>(nodes16, extra, gemb, exps, msgmaxbits, ut1, ub1,
                                       ut2, ub2, (float*)d_out, N);
}
